// CoDABiMambaDecoderLayer_34866544508976
// MI455X (gfx1250) — matrix operand order repaired and checked by reading, not compiled, not run on hardware
//
#include <hip/hip_runtime.h>
#include <hip/hip_bf16.h>
#include <stdint.h>

// ---------------------------------------------------------------------------
// CoDA BiMamba decoder layer for MI455X (gfx1250, wave32).
// GEMMs: bf16 operands in global memory (converted once), tiles DMA'd to LDS
// by the Tensor Data Mover (tensor_load_to_lds, TENSORcnt double-buffered),
// math on v_wmma_f32_16x16x32_bf16 with fp32 accumulation.
// ---------------------------------------------------------------------------

#define L_SEQ   2048
#define D_MODEL 1024
#define D_INNER 2048
#define D_STATE 16
#define DT_RANK 64
#define I_SIZE  4096
#define DBC_N   96   // DT_RANK + 2*D_STATE

typedef __bf16 bf16_t;
typedef bf16_t v16bf __attribute__((ext_vector_type(16)));
typedef bf16_t v8bf  __attribute__((ext_vector_type(8)));
typedef float  v8f   __attribute__((ext_vector_type(8)));
typedef uint32_t u32x4 __attribute__((ext_vector_type(4)));
typedef int      i32x8 __attribute__((ext_vector_type(8)));
typedef int      i32x4 __attribute__((ext_vector_type(4)));

#ifndef __has_builtin
#define __has_builtin(x) 0
#endif
#if __has_builtin(__builtin_amdgcn_tensor_load_to_lds) && \
    __has_builtin(__builtin_amdgcn_s_wait_tensorcnt)
#define USE_TDM 1
#else
#define USE_TDM 0
#endif

__device__ __forceinline__ bf16_t f2bf(float f) {
  union { float f; uint32_t u; } a; a.f = f;
  uint32_t r = a.u + 0x7fffu + ((a.u >> 16) & 1u);   // round-to-nearest-even
  uint16_t h = (uint16_t)(r >> 16);
  return __builtin_bit_cast(bf16_t, h);
}

__device__ __forceinline__ float sigmoidf_fast(float x) {
  return 1.0f / (1.0f + __expf(-x));
}

// ---------------------------------------------------------------------------
// GEMM: C[M,N](fp32) = A[M,K](bf16) * W[N,K](bf16)^T.
// 128x128 block, 256 threads = 8 wave32 (4x2), each wave 2x4 WMMA 16x16 tiles.
// ---------------------------------------------------------------------------
#define TM  128
#define TN  128
#define TK  32
#define LDT 40            // padded LDS row stride in bf16 (80 B)
#define TILE_B 10240u     // TM * LDT * 2 bytes (incl. trailing pad row space)
#define SMEM_B 40960u     // 2 buffers x (A tile + B tile)

#if USE_TDM
// Build a D# (ISA 8.3/8.4) for a 2-D bf16 tile: tile_dim0=TK (K elems),
// tile_dim1=TM rows, LDS padding 4 dwords after every 16 dwords -> 80B rows.
// rows_rem/k_rem bound the tensor so hardware zero-fills out-of-range rows.
// This toolchain exposes the 6-arg builtin:
//   (u32x4 g0, i32x8 g1, i32x4 g2, i32x4 g3, i32x8 extra, i32 cpol)
__device__ __forceinline__ void tdm_tile_bf16(uint32_t lds_off,
                                              const bf16_t* g,
                                              uint32_t rows_rem,
                                              uint32_t k_rem,
                                              uint32_t ld)
{
  u32x4 g0;
  g0[0] = 1u;                                   // count=1, user descriptor
  g0[1] = lds_off;                              // lds_addr
  uint64_t ga = (uint64_t)(uintptr_t)g;
  g0[2] = (uint32_t)ga;                         // global_addr[31:0]
  g0[3] = (uint32_t)((ga >> 32) & 0x01FFFFFFu)  // global_addr[56:32]
        | 0x80000000u;                          // type=2 ("image")
  i32x8 g1;
  g1[0] = (int)((1u << 16)    // data_size = 2 bytes
              | (1u << 20)    // pad_enable
              | (3u << 22)    // pad_interval: 16 dwords (64B)
              | (3u << 25));  // pad_amount:   4 dwords (16B)
  g1[1] = (int)((k_rem & 0xFFFFu) << 16);                      // tdim0 lo
  g1[2] = (int)(((k_rem >> 16) & 0xFFFFu)
              | ((rows_rem & 0xFFFFu) << 16));                 // tdim0 hi|tdim1 lo
  g1[3] = (int)(((rows_rem >> 16) & 0xFFFFu)
              | ((uint32_t)TK << 16));                         // tdim1 hi|tile_dim0
  g1[4] = (int)(uint32_t)TM;                                   // tile_dim1 (dim2=0)
  g1[5] = (int)ld;                                             // dim0_stride lo
  g1[6] = 0;                                                   // stride hi | d1s lo
  g1[7] = 0;
  i32x4 z4 = {0, 0, 0, 0};
  i32x8 z8 = {0, 0, 0, 0, 0, 0, 0, 0};
  __builtin_amdgcn_tensor_load_to_lds(g0, g1, z4, z4, z8, 0);
}
#endif

__global__ __launch_bounds__(256)
void gemm_bf16wmma(const bf16_t* __restrict__ A, int lda,
                   const bf16_t* __restrict__ W, int ldw,
                   float* __restrict__ C, int ldc,
                   int M, int N, int K)
{
  extern __shared__ char smem[];   // [buf0 A | buf0 B | buf1 A | buf1 B]

  const int tid   = threadIdx.x;
  const int lane  = tid & 31;
  const int wave  = tid >> 5;
  const int waveM = wave & 3;
  const int waveN = wave >> 2;
  const int half  = lane >> 4;
  const int lrow  = lane & 15;
  const int blockM = blockIdx.y * TM;
  const int blockN = blockIdx.x * TN;

  v8f acc[2][4];
#pragma unroll
  for (int mf = 0; mf < 2; ++mf)
#pragma unroll
    for (int nf = 0; nf < 4; ++nf)
      acc[mf][nf] = v8f{0.f, 0.f, 0.f, 0.f, 0.f, 0.f, 0.f, 0.f};

  const int nk = K / TK;

#if USE_TDM
  if (wave == 0) {
    tdm_tile_bf16(0u,      A + (size_t)blockM * lda, (uint32_t)(M - blockM),
                  (uint32_t)K, (uint32_t)lda);
    tdm_tile_bf16(TILE_B,  W + (size_t)blockN * ldw,
                  (uint32_t)(N > blockN ? N - blockN : 0),
                  (uint32_t)K, (uint32_t)ldw);
  }
#endif

  int cur = 0;
  for (int ik = 0; ik < nk; ++ik, cur ^= 1) {
    const int k0 = ik * TK;
#if USE_TDM
    if (wave == 0) {
      if (ik + 1 < nk) {
        const int kn = k0 + TK;
        const uint32_t nb = (uint32_t)(cur ^ 1) * (2u * TILE_B);
        tdm_tile_bf16(nb,          A + (size_t)blockM * lda + kn,
                      (uint32_t)(M - blockM), (uint32_t)(K - kn), (uint32_t)lda);
        tdm_tile_bf16(nb + TILE_B, W + (size_t)blockN * ldw + kn,
                      (uint32_t)(N > blockN ? N - blockN : 0),
                      (uint32_t)(K - kn), (uint32_t)ldw);
        __builtin_amdgcn_s_wait_tensorcnt(2);   // current buffer's pair done
      } else {
        __builtin_amdgcn_s_wait_tensorcnt(0);
      }
    }
    __syncthreads();
    const bf16_t* As = (const bf16_t*)(smem + (uint32_t)cur * (2u * TILE_B));
    const bf16_t* Bs = (const bf16_t*)(smem + (uint32_t)cur * (2u * TILE_B) + TILE_B);
#else
    bf16_t* AsW = (bf16_t*)smem;
    bf16_t* BsW = (bf16_t*)(smem + TILE_B);
    for (int i = tid; i < (TM * TK) / 8; i += 256) {
      int r = i >> 2, c = (i & 3) << 3;
      int gr = blockM + r;
      uint4 v = {0u, 0u, 0u, 0u};
      if (gr < M) v = *(const uint4*)(A + (size_t)gr * lda + k0 + c);
      *(uint4*)&AsW[r * LDT + c] = v;
    }
    for (int i = tid; i < (TN * TK) / 8; i += 256) {
      int r = i >> 2, c = (i & 3) << 3;
      int gr = blockN + r;
      uint4 v = {0u, 0u, 0u, 0u};
      if (gr < N) v = *(const uint4*)(W + (size_t)gr * ldw + k0 + c);
      *(uint4*)&BsW[r * LDT + c] = v;
    }
    __syncthreads();
    const bf16_t* As = AsW;
    const bf16_t* Bs = BsW;
#endif

    union FragU { v16bf v; v8bf h[2]; };
    v16bf afrag[2], bfrag[4];
    // A fragment (ISA 16-bit A 16x32): v[0..7]=K(8*half..), v[8..15]=K(16+8*half..)
#pragma unroll
    for (int mf = 0; mf < 2; ++mf) {
      int r = waveM * 32 + mf * 16 + lrow;
      FragU u;
      u.h[0] = *(const v8bf*)&As[r * LDT + half * 8];
      u.h[1] = *(const v8bf*)&As[r * LDT + 16 + half * 8];
      afrag[mf] = u.v;
    }
    // B fragment (dense 32x16): v[0..7]=K(8*half..), v[8..15]=K(16+8*half..)
#pragma unroll
    for (int nf = 0; nf < 4; ++nf) {
      int r = waveN * 64 + nf * 16 + lrow;
      FragU u;
      u.h[0] = *(const v8bf*)&Bs[r * LDT + half * 8];
      u.h[1] = *(const v8bf*)&Bs[r * LDT + 16 + half * 8];
      bfrag[nf] = u.v;
    }
#pragma unroll
    for (int mf = 0; mf < 2; ++mf)
#pragma unroll
      for (int nf = 0; nf < 4; ++nf)
        acc[mf][nf] = __builtin_amdgcn_wmma_f32_16x16x32_bf16(
            false, afrag[mf], false, bfrag[nf], (short)0, acc[mf][nf],
            false, false);
    __syncthreads();
  }

  // D layout: VGPR r -> M = r + 8*half, N = lrow
#pragma unroll
  for (int mf = 0; mf < 2; ++mf) {
    int rbase = blockM + waveM * 32 + mf * 16 + (half << 3);
#pragma unroll
    for (int nf = 0; nf < 4; ++nf) {
      int col = blockN + waveN * 64 + nf * 16 + lrow;
      if (col < N) {
#pragma unroll
        for (int r = 0; r < 8; ++r) {
          int row = rbase + r;
          if (row < M) C[(size_t)row * ldc + col] = acc[mf][nf][r];
        }
      }
    }
  }
}

// ---------------------------------------------------------------------------
// fp32 -> bf16 bulk convert (n multiple of 1024)
__global__ __launch_bounds__(256)
void f32_to_bf16_kernel(const float* __restrict__ src, bf16_t* __restrict__ dst)
{
  size_t i = ((size_t)blockIdx.x * 256 + threadIdx.x) * 4;
  float4 v = *(const float4*)(src + i);
  dst[i + 0] = f2bf(v.x); dst[i + 1] = f2bf(v.y);
  dst[i + 2] = f2bf(v.z); dst[i + 3] = f2bf(v.w);
}

// RMSNorm; optional mask, optional fp32 residual-add output, optional bf16 out
__global__ __launch_bounds__(256)
void rmsnorm_kernel(const float* __restrict__ x, const float* __restrict__ w,
                    const float* __restrict__ mask, const float* __restrict__ res,
                    float* __restrict__ outf, bf16_t* __restrict__ outb, int D)
{
  const int row = blockIdx.x;
  const int tid = threadIdx.x;
  const float* xr = x + (size_t)row * D;
  __shared__ float red[256];
  float s = 0.f;
  for (int i = tid; i < D; i += 256) { float v = xr[i]; s += v * v; }
  red[tid] = s;
  __syncthreads();
  for (int off = 128; off > 0; off >>= 1) {
    if (tid < off) red[tid] += red[tid + off];
    __syncthreads();
  }
  float rs = rsqrtf(red[0] / (float)D + 1e-6f);
  float m = mask ? mask[row] : 1.0f;
  for (int i = tid; i < D; i += 256) {
    float v = w[i] * (xr[i] * rs) * m;
    if (outf) outf[(size_t)row * D + i] =
        (res ? res[(size_t)row * D + i] : 0.f) + v;
    if (outb) outb[(size_t)row * D + i] = f2bf(v);
  }
}

// time-reverse rows (bf16): dst[t,:] = src[L-1-t,:]
__global__ __launch_bounds__(256)
void reverse_rows_kernel(const bf16_t* __restrict__ src, bf16_t* __restrict__ dst)
{
  int idx = blockIdx.x * 256 + threadIdx.x;
  int t = idx / D_MODEL;
  int c = idx - t * D_MODEL;
  dst[idx] = src[(size_t)(L_SEQ - 1 - t) * D_MODEL + c];
}

// depthwise causal conv (width 4) + bias + silu; xi = xz[:, :D_INNER]
// writes fp32 (for scan) and bf16 (for xproj GEMM)
__global__ __launch_bounds__(256)
void conv_silu_kernel(const float* __restrict__ xz, const float* __restrict__ w,
                      const float* __restrict__ b, float* __restrict__ xc,
                      bf16_t* __restrict__ xcb)
{
  int idx = blockIdx.x * 256 + threadIdx.x;
  int c = idx & (D_INNER - 1);
  int t = idx >> 11;
  float acc = b[c];
#pragma unroll
  for (int j = 0; j < 4; ++j) {
    int tt = t + j - 3;
    if (tt >= 0) acc += w[c * 4 + j] * xz[(size_t)tt * I_SIZE + c];
  }
  float v = acc * sigmoidf_fast(acc);
  xc[idx] = v;
  xcb[idx] = f2bf(v);
}

// Fused selective scan: h[16]/A[16] in VGPRs, B_t/C_t via LDS broadcast.
// dt = softplus(dtg + dt_b); y = (scan + u*Dp) * silu(z); out bf16.
__global__ __launch_bounds__(256)
void scan_kernel(const float* __restrict__ xc, const float* __restrict__ dtg,
                 const float* __restrict__ dbc, const float* __restrict__ xz,
                 const float* __restrict__ A_log, const float* __restrict__ dt_b,
                 const float* __restrict__ Dp, bf16_t* __restrict__ yout)
{
  const int d = blockIdx.x * 256 + threadIdx.x;
  __shared__ float sBC[32];
  float A[D_STATE], h[D_STATE];
#pragma unroll
  for (int n = 0; n < D_STATE; ++n) {
    A[n] = -__expf(A_log[(size_t)d * D_STATE + n]);
    h[n] = 0.f;
  }
  const float dpv  = Dp[d];
  const float bias = dt_b[d];

  for (int t = 0; t < L_SEQ; ++t) {
    if (threadIdx.x < 32) sBC[threadIdx.x] = dbc[(size_t)t * DBC_N + DT_RANK + threadIdx.x];
    __syncthreads();
    float dtv = dtg[(size_t)t * D_INNER + d] + bias;
    dtv = (dtv > 20.f) ? dtv : log1pf(__expf(dtv));
    float u  = xc[(size_t)t * D_INNER + d];
    float du = dtv * u;
    float y  = 0.f;
#pragma unroll
    for (int n = 0; n < D_STATE; ++n) {
      float dA = __expf(dtv * A[n]);
      h[n] = dA * h[n] + du * sBC[n];
      y += h[n] * sBC[16 + n];
    }
    float z = xz[(size_t)t * I_SIZE + D_INNER + d];
    yout[(size_t)t * D_INNER + d] = f2bf((y + u * dpv) * (z * sigmoidf_fast(z)));
    __syncthreads();
  }
}

// ysum[t,:] = yf[t,:] + yb[L-1-t,:]
__global__ __launch_bounds__(256)
void add_flip_kernel(const float* __restrict__ yf, const float* __restrict__ yb,
                     float* __restrict__ ysum)
{
  int idx = blockIdx.x * 256 + threadIdx.x;
  int t = idx / D_MODEL;
  int c = idx - t * D_MODEL;
  ysum[idx] = yf[idx] + yb[(size_t)(L_SEQ - 1 - t) * D_MODEL + c];
}

// act = silu(gate) * up  -> bf16 for the down GEMM
__global__ __launch_bounds__(256)
void silu_mul_kernel(const float* __restrict__ g, const float* __restrict__ u,
                     bf16_t* __restrict__ o)
{
  int idx = blockIdx.x * 256 + threadIdx.x;
  float gv = g[idx];
  o[idx] = f2bf(gv * sigmoidf_fast(gv) * u[idx]);
}

__global__ __launch_bounds__(256)
void add_kernel(const float* __restrict__ a, const float* __restrict__ b,
                float* __restrict__ o)
{
  int idx = blockIdx.x * 256 + threadIdx.x;
  o[idx] = a[idx] + b[idx];
}

// ---------------------------------------------------------------------------
extern "C" void kernel_launch(void* const* d_in, const int* in_sizes, int n_in,
                              void* d_out, int out_size, void* d_ws, size_t ws_size,
                              hipStream_t stream)
{
  (void)in_sizes; (void)n_in; (void)out_size; (void)ws_size;

  const float* hs      = (const float*)d_in[0];
  const float* amask   = (const float*)d_in[1];
  const float* mnorm_w = (const float*)d_in[2];
  const float* ln1_w   = (const float*)d_in[3];
  const float* ln2_w   = (const float*)d_in[4];
  const float* gate_w  = (const float*)d_in[5];
  const float* up_w    = (const float*)d_in[6];
  const float* down_w  = (const float*)d_in[7];

  struct DirW { const float *in_w, *conv_w, *conv_b, *xproj_w, *dt_w, *dt_b, *A_log, *Dp, *out_w; };
  DirW dw[2];
  for (int d = 0; d < 2; ++d) {
    int b = 8 + d * 9;
    dw[d].in_w    = (const float*)d_in[b + 0];
    dw[d].conv_w  = (const float*)d_in[b + 1];
    dw[d].conv_b  = (const float*)d_in[b + 2];
    dw[d].xproj_w = (const float*)d_in[b + 3];
    dw[d].dt_w    = (const float*)d_in[b + 4];
    dw[d].dt_b    = (const float*)d_in[b + 5];
    dw[d].A_log   = (const float*)d_in[b + 6];
    dw[d].Dp      = (const float*)d_in[b + 7];
    dw[d].out_w   = (const float*)d_in[b + 8];
  }

  // ---- workspace carve-up (~200 MB) ----
  char* ws = (char*)d_ws;
  size_t off = 0;
  auto alloc = [&](size_t bytes) -> char* {
    char* p = ws + off;
    off += (bytes + 255) & ~(size_t)255;
    return p;
  };
  // fp32 scratch
  float* xz   = (float*)alloc((size_t)L_SEQ * I_SIZE  * 4);  // MLP: gate
  float* xc   = (float*)alloc((size_t)L_SEQ * D_INNER * 4);  // MLP: up lo
  float* dtg  = (float*)alloc((size_t)L_SEQ * D_INNER * 4);  // MLP: up hi
  float* dbc  = (float*)alloc((size_t)L_SEQ * DBC_N   * 4);
  float* yfo  = (float*)alloc((size_t)L_SEQ * D_MODEL * 4);
  float* ybo  = (float*)alloc((size_t)L_SEQ * D_MODEL * 4);
  float* ysum = (float*)alloc((size_t)L_SEQ * D_MODEL * 4);
  float* xres = (float*)alloc((size_t)L_SEQ * D_MODEL * 4);
  float* mlp  = (float*)alloc((size_t)L_SEQ * D_MODEL * 4);
  float* gate = xz;
  float* up   = xc;     // xc+dtg contiguous 32MB
  // bf16 activations
  bf16_t* hnb   = (bf16_t*)alloc((size_t)L_SEQ * D_MODEL * 2);
  bf16_t* hrevb = (bf16_t*)alloc((size_t)L_SEQ * D_MODEL * 2);
  bf16_t* xcb   = (bf16_t*)alloc((size_t)L_SEQ * D_INNER * 2);
  bf16_t* dbcb  = (bf16_t*)alloc((size_t)L_SEQ * DBC_N   * 2);
  bf16_t* ydirb = (bf16_t*)alloc((size_t)L_SEQ * D_INNER * 2);
  bf16_t* h2b   = (bf16_t*)alloc((size_t)L_SEQ * D_MODEL * 2);
  bf16_t* actb  = (bf16_t*)alloc((size_t)L_SEQ * I_SIZE  * 2);
  // bf16 weights (converted once per launch)
  bf16_t* in_wb[2], *xprojb[2], *dtwb[2], *outwb[2];
  for (int d = 0; d < 2; ++d) {
    in_wb[d]  = (bf16_t*)alloc((size_t)I_SIZE * D_MODEL * 2);
    xprojb[d] = (bf16_t*)alloc((size_t)DBC_N * D_INNER * 2);
    dtwb[d]   = (bf16_t*)alloc((size_t)D_INNER * DT_RANK * 2);
    outwb[d]  = (bf16_t*)alloc((size_t)D_MODEL * D_INNER * 2);
  }
  bf16_t* gatewb = (bf16_t*)alloc((size_t)I_SIZE * D_MODEL * 2);
  bf16_t* upwb   = (bf16_t*)alloc((size_t)I_SIZE * D_MODEL * 2);
  bf16_t* downwb = (bf16_t*)alloc((size_t)D_MODEL * I_SIZE * 2);

  const dim3 blk(256);
  const int ELW = 256;
  auto cvt = [&](const float* s, bf16_t* dst, size_t n) {
    f32_to_bf16_kernel<<<(unsigned)(n / 1024), blk, 0, stream>>>(s, dst);
  };

  // weight conversion (one pass each; reused 2048x by the GEMMs)
  for (int d = 0; d < 2; ++d) {
    cvt(dw[d].in_w,    in_wb[d],  (size_t)I_SIZE * D_MODEL);
    cvt(dw[d].xproj_w, xprojb[d], (size_t)DBC_N * D_INNER);
    cvt(dw[d].dt_w,    dtwb[d],   (size_t)D_INNER * DT_RANK);
    cvt(dw[d].out_w,   outwb[d],  (size_t)D_MODEL * D_INNER);
  }
  cvt(gate_w, gatewb, (size_t)I_SIZE * D_MODEL);
  cvt(up_w,   upwb,   (size_t)I_SIZE * D_MODEL);
  cvt(down_w, downwb, (size_t)D_MODEL * I_SIZE);

  // h = rmsnorm(x, ln1) * mask  (bf16 for GEMM A); reversed copy for bwd dir
  rmsnorm_kernel<<<L_SEQ, blk, 0, stream>>>(hs, ln1_w, amask, nullptr,
                                            nullptr, hnb, D_MODEL);
  reverse_rows_kernel<<<(L_SEQ * D_MODEL) / ELW, blk, 0, stream>>>(hnb, hrevb);

  const bf16_t* dirin[2]  = {hnb, hrevb};
  float*        dirout[2] = {yfo, ybo};
  for (int d = 0; d < 2; ++d) {
    // xz = h @ in_w^T  [2048,4096,K=1024]
    gemm_bf16wmma<<<dim3(I_SIZE / TN, L_SEQ / TM), blk, SMEM_B, stream>>>(
        dirin[d], D_MODEL, in_wb[d], D_MODEL, xz, I_SIZE, L_SEQ, I_SIZE, D_MODEL);
    // xc = silu(causal_conv4(xi) + b)
    conv_silu_kernel<<<(L_SEQ * D_INNER) / ELW, blk, 0, stream>>>(
        xz, dw[d].conv_w, dw[d].conv_b, xc, xcb);
    // dbc = xc @ xproj^T  [2048,96,K=2048] (OOB rows zero-filled by TDM)
    gemm_bf16wmma<<<dim3((DBC_N + TN - 1) / TN, L_SEQ / TM), blk, SMEM_B, stream>>>(
        xcb, D_INNER, xprojb[d], D_INNER, dbc, DBC_N, L_SEQ, DBC_N, D_INNER);
    cvt(dbc, dbcb, (size_t)L_SEQ * DBC_N);
    // dtg = dt_raw @ dt_w^T  [2048,2048,K=64]  (dt_raw = dbc[:, :64])
    gemm_bf16wmma<<<dim3(D_INNER / TN, L_SEQ / TM), blk, SMEM_B, stream>>>(
        dbcb, DBC_N, dtwb[d], DT_RANK, dtg, D_INNER, L_SEQ, D_INNER, DT_RANK);
    // fused selective scan (+softplus, +Dp, *silu(z)) -> bf16
    scan_kernel<<<D_INNER / 256, blk, 0, stream>>>(
        xc, dtg, dbc, xz, dw[d].A_log, dw[d].dt_b, dw[d].Dp, ydirb);
    // y_dir = y @ out_w^T  [2048,1024,K=2048]
    gemm_bf16wmma<<<dim3(D_MODEL / TN, L_SEQ / TM), blk, SMEM_B, stream>>>(
        ydirb, D_INNER, outwb[d], D_INNER, dirout[d], D_MODEL, L_SEQ, D_MODEL, D_INNER);
  }

  // x = x + rmsnorm(y_f + flip(y_b), mnorm)
  add_flip_kernel<<<(L_SEQ * D_MODEL) / ELW, blk, 0, stream>>>(yfo, ybo, ysum);
  rmsnorm_kernel<<<L_SEQ, blk, 0, stream>>>(ysum, mnorm_w, nullptr, hs,
                                            xres, nullptr, D_MODEL);

  // MLP: silu(h2 @ gate^T) * (h2 @ up^T) @ down^T
  rmsnorm_kernel<<<L_SEQ, blk, 0, stream>>>(xres, ln2_w, nullptr, nullptr,
                                            nullptr, h2b, D_MODEL);
  gemm_bf16wmma<<<dim3(I_SIZE / TN, L_SEQ / TM), blk, SMEM_B, stream>>>(
      h2b, D_MODEL, gatewb, D_MODEL, gate, I_SIZE, L_SEQ, I_SIZE, D_MODEL);
  gemm_bf16wmma<<<dim3(I_SIZE / TN, L_SEQ / TM), blk, SMEM_B, stream>>>(
      h2b, D_MODEL, upwb, D_MODEL, up, I_SIZE, L_SEQ, I_SIZE, D_MODEL);
  silu_mul_kernel<<<(L_SEQ * I_SIZE) / ELW, blk, 0, stream>>>(gate, up, actb);
  gemm_bf16wmma<<<dim3(D_MODEL / TN, L_SEQ / TM), blk, SMEM_B, stream>>>(
      actb, I_SIZE, downwb, I_SIZE, mlp, D_MODEL, L_SEQ, D_MODEL, I_SIZE);

  add_kernel<<<(L_SEQ * D_MODEL) / ELW, blk, 0, stream>>>(xres, mlp, (float*)d_out);
}
